// Spann3R_2997887172779
// MI455X (gfx1250) — matrix operand order repaired and checked by reading, not compiled, not run on hardware
//
#include <hip/hip_runtime.h>

// ---------------------------------------------------------------------------
// CDNA5 (gfx1250) fused cross-attention read:
//   q = LN(feat)*scale; k = LN(mem_k); v = LN(mem_v)
//   S = (q k^T) * mem_c          (scale pre-folded into q)
//   attn = renorm(threshold(softmax(S)))
//   out = attn v + feat ; mem_attn += colsum(attn)
// bf16 WMMA (v_wmma_f32_16x16x32_bf16) for both GEMMs, fp32 accumulate.
// GEMM inner loops are register double-buffered; schedule pinned with
// __builtin_amdgcn_sched_barrier so next-step loads issue before current
// WMMAs (per-wave latency hiding, not just wave-level).
// ---------------------------------------------------------------------------

typedef __attribute__((ext_vector_type(16))) __bf16 v16bf;
typedef __attribute__((ext_vector_type(4)))  __bf16 v4bf;
typedef __attribute__((ext_vector_type(8)))  float  v8f;
typedef __attribute__((ext_vector_type(4)))  float  f32x4;
typedef __attribute__((ext_vector_type(4)))  unsigned int u32x4;

#define LN_EPS      1e-5f
#define ATTN_THRESH 0.0005f

#if defined(__has_builtin)
#  if __has_builtin(__builtin_amdgcn_sched_barrier)
#    define SCHED_FENCE() __builtin_amdgcn_sched_barrier(0)
#  endif
#endif
#ifndef SCHED_FENCE
#  define SCHED_FENCE()
#endif

// Build a v16bf WMMA operand from two contiguous 16-byte chunks.
static __device__ __forceinline__ v16bf load2x16(const __bf16* p0, const __bf16* p1) {
  union { v16bf v; u32x4 q[2]; } t;
  t.q[0] = *(const u32x4*)p0;
  t.q[1] = *(const u32x4*)p1;
  return t.v;
}

static __device__ __forceinline__ v8f wmma_bf16(v16bf a, v16bf b, v8f c) {
  return __builtin_amdgcn_wmma_f32_16x16x32_bf16(false, a, false, b, (short)0, c,
                                                 false, false);
}

// ---------------------------------------------------------------------------
// LayerNorm over C=1024, output bf16 scaled by out_scale.
// One 256-thread block per row.
// ---------------------------------------------------------------------------
__global__ __launch_bounds__(256)
void ln_bf16_kernel(const float* __restrict__ x, const float* __restrict__ gamma,
                    const float* __restrict__ beta, __bf16* __restrict__ y,
                    float out_scale) {
  const int C = 1024;
  const int row = blockIdx.x, tid = threadIdx.x;
  const float* xr = x + (size_t)row * C;
  f32x4 v = *(const f32x4*)(xr + (tid << 2));
  float s  = v[0] + v[1] + v[2] + v[3];
  float ss = v[0]*v[0] + v[1]*v[1] + v[2]*v[2] + v[3]*v[3];
  __shared__ float r0[256], r1[256];
  r0[tid] = s; r1[tid] = ss;
  __syncthreads();
  for (int off = 128; off > 0; off >>= 1) {
    if (tid < off) { r0[tid] += r0[tid + off]; r1[tid] += r1[tid + off]; }
    __syncthreads();
  }
  const float mu  = r0[0] * (1.0f / C);
  const float var = r1[0] * (1.0f / C) - mu * mu;
  const float rs  = rsqrtf(var + LN_EPS);
  f32x4 g = *(const f32x4*)(gamma + (tid << 2));
  f32x4 b = *(const f32x4*)(beta  + (tid << 2));
  v4bf o;
#pragma unroll
  for (int i = 0; i < 4; i++)
    o[i] = (__bf16)(((v[i] - mu) * rs * g[i] + b[i]) * out_scale);
  *(v4bf*)(y + (size_t)row * C + (tid << 2)) = o;
}

// ---------------------------------------------------------------------------
// bf16 transpose [X,C] -> [C,X] per batch, 32x32 LDS tiles.
// ---------------------------------------------------------------------------
__global__ __launch_bounds__(256)
void transpose_bf16_kernel(const __bf16* __restrict__ in, __bf16* __restrict__ out) {
  const int X = 8192, C = 1024;
  __shared__ __bf16 tile[32][33];
  const int b  = blockIdx.z;
  const int x0 = blockIdx.x << 5, c0 = blockIdx.y << 5;
  const __bf16* ip = in  + (size_t)b * X * C;
  __bf16*       op = out + (size_t)b * C * X;
#pragma unroll
  for (int i = 0; i < 4; i++) {
    const int x = x0 + threadIdx.y + i * 8;
    tile[threadIdx.y + i * 8][threadIdx.x] = ip[(size_t)x * C + c0 + threadIdx.x];
  }
  __syncthreads();
#pragma unroll
  for (int i = 0; i < 4; i++) {
    const int c = c0 + threadIdx.y + i * 8;
    op[(size_t)c * X + x0 + threadIdx.x] = tile[threadIdx.x][threadIdx.y + i * 8];
  }
}

// ---------------------------------------------------------------------------
// GEMM1: S[b,p,x] = mem_c[b,x] * sum_c q[b,p,c] * k[b,x,c]   (q pre-scaled)
// Wave tile 16(M=P) x 64(N=X); 4 waves/block cover 16x256; K=C in steps of 32.
// A layout (16x32 bf16): K = 16*(lane/16)+h (contiguous).
// B layout (32x16 bf16): N = lane%16 = k-row, K = 16*(lane/16)+h (contiguous).
// Double-buffered; sched_barrier pins loads(k+1) ahead of WMMAs(k).
// ---------------------------------------------------------------------------
__global__ __launch_bounds__(128)
void gemm_qk_kernel(const __bf16* __restrict__ qbf, const __bf16* __restrict__ kbf,
                    const float* __restrict__ memc, float* __restrict__ S) {
  const int P = 576, X = 8192, C = 1024;
  const int b = blockIdx.z, pt = blockIdx.y, xt = blockIdx.x;
  const int wave = threadIdx.x >> 5, lane = threadIdx.x & 31;
  const int l16 = lane & 15, lh = lane >> 4;

  const __bf16* qrow = qbf + (size_t)(b * P + pt * 16 + l16) * C + 16 * lh;
  const int xw = xt * 256 + wave * 64;
  const __bf16* krow = kbf + (size_t)(b * X + xw + l16) * C + 16 * lh;

  v8f acc[4] = {};
  // --- prologue: load K-step 0 ---
  v16bf A = load2x16(qrow, qrow + 8);
  v16bf Bv[4];
#pragma unroll
  for (int j = 0; j < 4; j++) {
    const __bf16* kr = krow + (size_t)(j * 16) * C;
    Bv[j] = load2x16(kr, kr + 8);
  }
  // --- steady state: issue K-step k loads, then K-step k-32 WMMAs ---
#pragma unroll 2
  for (int k = 32; k < C; k += 32) {
    v16bf An = load2x16(qrow + k, qrow + k + 8);
    v16bf Bn[4];
#pragma unroll
    for (int j = 0; j < 4; j++) {
      const __bf16* kr = krow + (size_t)(j * 16) * C + k;
      Bn[j] = load2x16(kr, kr + 8);
    }
    SCHED_FENCE();          // loads(k) must issue before WMMAs(k-32)
#pragma unroll
    for (int j = 0; j < 4; j++) acc[j] = wmma_bf16(A, Bv[j], acc[j]);
    SCHED_FENCE();          // keep buffer rotation out of the compute group
    A = An;
#pragma unroll
    for (int j = 0; j < 4; j++) Bv[j] = Bn[j];
  }
  // --- epilogue ---
#pragma unroll
  for (int j = 0; j < 4; j++) acc[j] = wmma_bf16(A, Bv[j], acc[j]);

  const float* cc = memc + (size_t)b * X;
  float* sb = S + (size_t)(b * P + pt * 16) * X;
#pragma unroll
  for (int j = 0; j < 4; j++) {
    const int xcol = xw + j * 16 + l16;
    const float cv = cc[xcol];
#pragma unroll
    for (int g = 0; g < 8; g++)
      sb[(size_t)(g + 8 * lh) * X + xcol] = acc[j][g] * cv;   // D: M=g+8*lh, N=l16
  }
}

// ---------------------------------------------------------------------------
// Softmax + threshold + renorm over X=8192. One block per (b,p) row.
// Rewrites attn as bf16 in place over its own fp32 row (pitch 2X bf16 elems).
// ---------------------------------------------------------------------------
__global__ __launch_bounds__(256)
void softmax_kernel(float* __restrict__ S) {
  const int X = 8192;
  const int row = blockIdx.x, tid = threadIdx.x;
  float* sr = S + (size_t)row * X;
  float vals[32];
  float m = -3.0e38f;
#pragma unroll
  for (int i = 0; i < 32; i++) { vals[i] = sr[tid + (i << 8)]; m = fmaxf(m, vals[i]); }
  __shared__ float red[256];
  red[tid] = m; __syncthreads();
  for (int off = 128; off > 0; off >>= 1) {
    if (tid < off) red[tid] = fmaxf(red[tid], red[tid + off]);
    __syncthreads();
  }
  m = red[0]; __syncthreads();

  float sum = 0.f;
#pragma unroll
  for (int i = 0; i < 32; i++) { vals[i] = __expf(vals[i] - m); sum += vals[i]; }
  red[tid] = sum; __syncthreads();
  for (int off = 128; off > 0; off >>= 1) {
    if (tid < off) red[tid] += red[tid + off];
    __syncthreads();
  }
  const float inv = 1.0f / red[0]; __syncthreads();

  float ks = 0.f;
#pragma unroll
  for (int i = 0; i < 32; i++) {
    float p = vals[i] * inv;
    p = (p < ATTN_THRESH) ? 0.0f : p;
    vals[i] = p; ks += p;
  }
  red[tid] = ks; __syncthreads();
  for (int off = 128; off > 0; off >>= 1) {
    if (tid < off) red[tid] += red[tid + off];
    __syncthreads();
  }
  const float rn = 1.0f / red[0];

  __bf16* ar = (__bf16*)sr;   // in-place; bf16 row uses first half of fp32 row
#pragma unroll
  for (int i = 0; i < 32; i++) ar[tid + (i << 8)] = (__bf16)(vals[i] * rn);
}

// ---------------------------------------------------------------------------
// mem_attn_new[b,x] = mem_attn[b,x] + sum_p attn[b,p,x]   (no atomics)
// ---------------------------------------------------------------------------
__global__ __launch_bounds__(256)
void colsum_kernel(const __bf16* __restrict__ attn, const float* __restrict__ memattn,
                   float* __restrict__ outattn) {
  const int P = 576, X = 8192;
  const size_t AP = 2 * (size_t)X;            // bf16 row pitch
  const int idx = blockIdx.x * 256 + threadIdx.x;   // b*X + x
  const int b = idx >> 13;
  const int x = idx & (X - 1);
  const __bf16* col = attn + (size_t)(b * P) * AP + x;
  float s = 0.f;
  for (int p = 0; p < P; p++) s += (float)col[(size_t)p * AP];
  outattn[idx] = memattn[idx] + s;
}

// ---------------------------------------------------------------------------
// GEMM2: out[b,p,c] = sum_x attn[b,p,x] * v[b,x,c] + feat[b,p,c]
// B operand reads vT [C,X] contiguously (lane = column c of v = row of vT).
// Same double-buffered + sched_barrier structure; K=X (256 iterations).
// ---------------------------------------------------------------------------
__global__ __launch_bounds__(128)
void gemm_av_kernel(const __bf16* __restrict__ attn, const __bf16* __restrict__ vT,
                    const float* __restrict__ feat, float* __restrict__ out) {
  const int P = 576, X = 8192, C = 1024;
  const size_t AP = 2 * (size_t)X;
  const int b = blockIdx.z, pt = blockIdx.y, ct = blockIdx.x;
  const int wave = threadIdx.x >> 5, lane = threadIdx.x & 31;
  const int l16 = lane & 15, lh = lane >> 4;

  const __bf16* arow = attn + (size_t)(b * P + pt * 16 + l16) * AP + 16 * lh;
  const int cw = ct * 256 + wave * 64;
  const __bf16* vrow = vT + (size_t)(b * C + cw + l16) * X + 16 * lh;

  v8f acc[4] = {};
  // --- prologue ---
  v16bf A = load2x16(arow, arow + 8);
  v16bf Bv[4];
#pragma unroll
  for (int j = 0; j < 4; j++) {
    const __bf16* vr = vrow + (size_t)(j * 16) * X;
    Bv[j] = load2x16(vr, vr + 8);
  }
  // --- steady state ---
#pragma unroll 2
  for (int k = 32; k < X; k += 32) {
    v16bf An = load2x16(arow + k, arow + k + 8);
    v16bf Bn[4];
#pragma unroll
    for (int j = 0; j < 4; j++) {
      const __bf16* vr = vrow + (size_t)(j * 16) * X + k;
      Bn[j] = load2x16(vr, vr + 8);
    }
    SCHED_FENCE();          // loads(k) must issue before WMMAs(k-32)
#pragma unroll
    for (int j = 0; j < 4; j++) acc[j] = wmma_bf16(A, Bv[j], acc[j]);
    SCHED_FENCE();
    A = An;
#pragma unroll
    for (int j = 0; j < 4; j++) Bv[j] = Bn[j];
  }
  // --- epilogue ---
#pragma unroll
  for (int j = 0; j < 4; j++) acc[j] = wmma_bf16(A, Bv[j], acc[j]);

#pragma unroll
  for (int j = 0; j < 4; j++) {
    const int ccol = cw + j * 16 + l16;
#pragma unroll
    for (int g = 0; g < 8; g++) {
      const size_t idx = (size_t)(b * P + pt * 16 + g + 8 * lh) * C + ccol;
      out[idx] = acc[j][g] + feat[idx];
    }
  }
}

// ---------------------------------------------------------------------------
extern "C" void kernel_launch(void* const* d_in, const int* in_sizes, int n_in,
                              void* d_out, int out_size, void* d_ws, size_t ws_size,
                              hipStream_t stream) {
  (void)in_sizes; (void)n_in; (void)out_size; (void)ws_size;
  const int B = 2, P = 576, X = 8192, C = 1024;

  const float* feat     = (const float*)d_in[0];
  const float* mem_k    = (const float*)d_in[1];
  const float* mem_v    = (const float*)d_in[2];
  const float* mem_c    = (const float*)d_in[3];
  const float* mem_attn = (const float*)d_in[4];
  const float* gq = (const float*)d_in[5], *bq = (const float*)d_in[6];
  const float* gk = (const float*)d_in[7], *bk = (const float*)d_in[8];
  const float* gv = (const float*)d_in[9], *bv = (const float*)d_in[10];

  // Workspace layout (all 256B-aligned offsets):
  //   qbf [B,P,C] bf16 | kbf [B,X,C] bf16 | vbf [B,X,C] bf16 |
  //   vT [B,C,X] bf16  | S [B,P,X] fp32 (attn bf16 aliases S, pitch 2X)
  char* ws = (char*)d_ws;
  __bf16* qbf = (__bf16*)ws;  ws += (size_t)B * P * C * 2;
  __bf16* kbf = (__bf16*)ws;  ws += (size_t)B * X * C * 2;
  __bf16* vbf = (__bf16*)ws;  ws += (size_t)B * X * C * 2;
  __bf16* vT  = (__bf16*)ws;  ws += (size_t)B * X * C * 2;
  float*  S   = (float*)ws;

  float* out_feat = (float*)d_out;                       // [B,P,C]
  float* out_attn = (float*)d_out + (size_t)B * P * C;   // [B,X,1]

  // 1) LayerNorms -> bf16 (1/sqrt(C) folded into q)
  ln_bf16_kernel<<<B * P, 256, 0, stream>>>(feat,  gq, bq, qbf, 0.03125f);
  ln_bf16_kernel<<<B * X, 256, 0, stream>>>(mem_k, gk, bk, kbf, 1.0f);
  ln_bf16_kernel<<<B * X, 256, 0, stream>>>(mem_v, gv, bv, vbf, 1.0f);

  // 2) v -> vT for contiguous WMMA B-operand loads in GEMM2
  transpose_bf16_kernel<<<dim3(X / 32, C / 32, B), dim3(32, 8), 0, stream>>>(vbf, vT);

  // 3) S = mem_c * (q k^T)
  gemm_qk_kernel<<<dim3(X / 256, P / 16, B), 128, 0, stream>>>(qbf, kbf, mem_c, S);

  // 4) softmax + threshold + renorm; attn bf16 written in place over S
  softmax_kernel<<<B * P, 256, 0, stream>>>(S);

  // 5) mem_attn update (deterministic column reduction)
  colsum_kernel<<<(B * X) / 256, 256, 0, stream>>>((const __bf16*)S, mem_attn, out_attn);

  // 6) out = attn v + feat
  gemm_av_kernel<<<dim3(C / 256, P / 16, B), 128, 0, stream>>>((const __bf16*)S, vT, feat, out_feat);
}
